// RouteAModel_39290360823924
// MI455X (gfx1250) — matrix operand order repaired and checked by reading, not compiled, not run on hardware
//
#include <hip/hip_runtime.h>
#include <math.h>

// ---------------------------------------------------------------------------
// Problem constants (match reference setup_inputs)
// ---------------------------------------------------------------------------
#define BB   8
#define HH   384
#define WW   512
#define HW   (HH * WW)          // 196608
#define BHW  (BB * HW)          // 1572864
#define H2   (HH / 2)
#define W2   (WW / 2)
#define H4   (HH / 4)
#define W4   (WW / 4)

#define TEMP_       30.0f
#define HARD_Z_EPS_ 0.001f
#define EPS_        1e-06f

typedef _Float16 half_t;
typedef __attribute__((ext_vector_type(16))) _Float16 v16h;
typedef __attribute__((ext_vector_type(8)))  _Float16 v8h;
typedef __attribute__((ext_vector_type(8)))  float    v8f;

// ---------------------------------------------------------------------------
// Per-batch 3x3 inverse of Ks
// ---------------------------------------------------------------------------
__global__ void kinv_kernel(const float* __restrict__ Ks, float* __restrict__ kinv) {
    int b = threadIdx.x + blockIdx.x * blockDim.x;
    if (b >= BB) return;
    const float* k = Ks + b * 9;
    float a = k[0], b1 = k[1], c = k[2];
    float d = k[3], e  = k[4], f = k[5];
    float g = k[6], h  = k[7], i = k[8];
    float A  =  (e * i - f * h);
    float Bc = -(d * i - f * g);
    float C  =  (d * h - e * g);
    float det = a * A + b1 * Bc + c * C;
    float inv = 1.0f / det;
    float* o = kinv + b * 9;
    o[0] =  A * inv;   o[1] = -(b1 * i - c * h) * inv; o[2] =  (b1 * f - c * e) * inv;
    o[3] =  Bc * inv;  o[4] =  (a * i - c * g) * inv;  o[5] = -(a * f - c * d) * inv;
    o[6] =  C * inv;   o[7] = -(a * h - b1 * g) * inv; o[8] =  (a * e - b1 * d) * inv;
}

// ---------------------------------------------------------------------------
// Init: zbuf = +inf, acc = 0, zmax = 0 (runs every call: graph-replay safe)
// ---------------------------------------------------------------------------
__global__ void init_kernel(float* __restrict__ zbuf, float* __restrict__ acc,
                            float* __restrict__ zmax) {
    int i = threadIdx.x + blockIdx.x * blockDim.x;
    if (i < BB) zmax[i] = 0.0f;
    if (i >= BHW) return;
    zbuf[i] = __uint_as_float(0x7f800000u);   // +inf
    acc[0 * BHW + i] = 0.0f;
    acc[1 * BHW + i] = 0.0f;
    acc[2 * BHW + i] = 0.0f;
    acc[3 * BHW + i] = 0.0f;
}

// ---------------------------------------------------------------------------
// Projection: xt, yt, z; grid_geo and zt straight into d_out; zmax atomic
// ---------------------------------------------------------------------------
__global__ void project_kernel(const float* __restrict__ Ds,
                               const float* __restrict__ Kt,
                               const float* __restrict__ dT,
                               const float* __restrict__ kinv,
                               float* __restrict__ xtb, float* __restrict__ ytb,
                               float* __restrict__ zb,  float* __restrict__ zmax,
                               float* __restrict__ out_gg, float* __restrict__ out_zt) {
    int idx = threadIdx.x + blockIdx.x * blockDim.x;
    if (idx >= BHW) return;
    int b  = idx / HW;
    int hw = idx - b * HW;
    int y  = hw / WW;
    int x  = hw - y * WW;

    float d = fmaxf(Ds[idx], 0.001f);
    const float* Ki = kinv + b * 9;
    float px = (float)x, py = (float)y;
    float cx = (Ki[0] * px + Ki[1] * py + Ki[2]) * d;
    float cy = (Ki[3] * px + Ki[4] * py + Ki[5]) * d;
    float cz = (Ki[6] * px + Ki[7] * py + Ki[8]) * d;
    const float* T = dT + b * 16;
    float X = T[0] * cx + T[1] * cy + T[2]  * cz + T[3];
    float Y = T[4] * cx + T[5] * cy + T[6]  * cz + T[7];
    float Z = T[8] * cx + T[9] * cy + T[10] * cz + T[11];
    const float* K = Kt + b * 9;
    float p0 = K[0] * X + K[1] * Y + K[2] * Z;
    float p1 = K[3] * X + K[4] * Y + K[5] * Z;
    float p2 = K[6] * X + K[7] * Y + K[8] * Z;
    float zc = fmaxf(p2, EPS_);
    float xt = p0 / zc, yt = p1 / zc;

    xtb[idx] = xt; ytb[idx] = yt; zb[idx] = p2;
    out_gg[idx * 2 + 0] = (2.0f * xt + 1.0f) / (float)WW - 1.0f;
    out_gg[idx * 2 + 1] = (2.0f * yt + 1.0f) / (float)HH - 1.0f;
    out_zt[idx] = p2;
    if (p2 > EPS_)
        atomicMax((unsigned int*)&zmax[b], __float_as_uint(p2));
}

// ---------------------------------------------------------------------------
// Z-buffer min pass (float-as-uint atomicMin valid for positive floats)
// ---------------------------------------------------------------------------
__global__ void zbuf_kernel(const float* __restrict__ xtb, const float* __restrict__ ytb,
                            const float* __restrict__ zb, float* __restrict__ zbuf) {
    int idx = threadIdx.x + blockIdx.x * blockDim.x;
    if (idx >= BHW) return;
    float z = zb[idx];
    if (!(z > EPS_)) return;
    int b = idx / HW;
    float xt = xtb[idx], yt = ytb[idx];
    float x0 = floorf(xt), y0 = floorf(yt);
    #pragma unroll
    for (int c = 0; c < 4; ++c) {
        float cxf = x0 + (float)(c & 1);
        float cyf = y0 + (float)(c >> 1);
        float bw = (1.0f - fabsf(xt - cxf)) * (1.0f - fabsf(yt - cyf));
        if (cxf >= 0.0f && cxf < (float)WW && cyf >= 0.0f && cyf < (float)HH && bw > 0.0f) {
            int flat = b * HW + (int)cyf * WW + (int)cxf;
            atomicMin((unsigned int*)&zbuf[flat], __float_as_uint(z));
        }
    }
}

// ---------------------------------------------------------------------------
// Weighted scatter-add pass
// ---------------------------------------------------------------------------
__global__ void splat_kernel(const float* __restrict__ Is,
                             const float* __restrict__ xtb, const float* __restrict__ ytb,
                             const float* __restrict__ zb, const float* __restrict__ zbuf,
                             const float* __restrict__ zmax, float* __restrict__ acc) {
    int idx = threadIdx.x + blockIdx.x * blockDim.x;
    if (idx >= BHW) return;
    float z = zb[idx];
    if (!(z > EPS_)) return;
    int b  = idx / HW;
    int hw = idx - b * HW;
    float zm = zmax[b] + EPS_;
    float w_imp = __expf(-TEMP_ * z / zm);
    float f0 = Is[(b * 3 + 0) * HW + hw];
    float f1 = Is[(b * 3 + 1) * HW + hw];
    float f2 = Is[(b * 3 + 2) * HW + hw];
    float xt = xtb[idx], yt = ytb[idx];
    float x0 = floorf(xt), y0 = floorf(yt);
    #pragma unroll
    for (int c = 0; c < 4; ++c) {
        float cxf = x0 + (float)(c & 1);
        float cyf = y0 + (float)(c >> 1);
        float bw = (1.0f - fabsf(xt - cxf)) * (1.0f - fabsf(yt - cyf));
        if (cxf >= 0.0f && cxf < (float)WW && cyf >= 0.0f && cyf < (float)HH && bw > 0.0f) {
            int flat = b * HW + (int)cyf * WW + (int)cxf;
            if (z <= zbuf[flat] + HARD_Z_EPS_) {
                float wt = bw * w_imp;
                atomicAdd(&acc[0 * BHW + flat], wt * f0);
                atomicAdd(&acc[1 * BHW + flat], wt * f1);
                atomicAdd(&acc[2 * BHW + flat], wt * f2);
                atomicAdd(&acc[3 * BHW + flat], wt);
            }
        }
    }
}

// ---------------------------------------------------------------------------
// Normalize: I_warp and V straight into d_out
// ---------------------------------------------------------------------------
__global__ void normalize_kernel(const float* __restrict__ acc,
                                 float* __restrict__ out_iw, float* __restrict__ out_v) {
    int idx = threadIdx.x + blockIdx.x * blockDim.x;
    if (idx >= BHW) return;
    int b  = idx / HW;
    int hw = idx - b * HW;
    float den = acc[3 * BHW + idx];
    float inv = 1.0f / fmaxf(den, EPS_);
    bool ok = den > EPS_;
    #pragma unroll
    for (int c = 0; c < 3; ++c) {
        float n = acc[c * BHW + idx];
        out_iw[(b * 3 + c) * HW + hw] = ok ? n * inv : 0.0f;
    }
    out_v[idx] = fminf(fmaxf(den, 0.0f), 1.0f);
}

// ---------------------------------------------------------------------------
// 5-channel concat: [a(3ch), b(1ch), c(1ch)] -> out (B,5,H,W)
// ---------------------------------------------------------------------------
__global__ void concat5_kernel(const float* __restrict__ a3, const float* __restrict__ b1,
                               const float* __restrict__ c1, float* __restrict__ out5) {
    int idx = threadIdx.x + blockIdx.x * blockDim.x;
    if (idx >= BHW) return;
    int b  = idx / HW;
    int hw = idx - b * HW;
    out5[(b * 5 + 0) * HW + hw] = a3[(b * 3 + 0) * HW + hw];
    out5[(b * 5 + 1) * HW + hw] = a3[(b * 3 + 1) * HW + hw];
    out5[(b * 5 + 2) * HW + hw] = a3[(b * 3 + 2) * HW + hw];
    out5[(b * 5 + 3) * HW + hw] = b1[idx];
    out5[(b * 5 + 4) * HW + hw] = c1[idx];
}

// ---------------------------------------------------------------------------
// Implicit-GEMM 3x3 conv via v_wmma_f32_16x16x32_f16, tap-major with LDS
// staging.  Block = 128 threads (4 waves) -> 64 output cols x 16 out channels.
// Per 32-channel chunk: stage input patch (3 x COLS x 32, f16, zero-padded)
// and weight tile (9 taps x 16 cout x 32 cin, f16) into LDS, then 9 unrolled
// taps, each = 2+2 ds_load_b128 + 1 WMMA.  Up to 2 input channel-segments
// (virtual concat) with 32-aligned boundary; seg0 optionally 2x-upsampled.
// act: 0 = none, 1 = relu, 2 = flow head (tanh ch0/1 -> out, sigmoid -> out2)
// ---------------------------------------------------------------------------
template <int STRIDE>
__global__ __launch_bounds__(128) void conv3x3_wmma(
        const float* __restrict__ s0, int c0, int u0,
        const float* __restrict__ s1, int u1,
        const float* __restrict__ wgt, const float* __restrict__ bias,
        float* __restrict__ out, float* __restrict__ out2,
        int Cin, int Cout, int Hi, int Wi, int Ho, int Wo, int act) {
    constexpr int PAD  = (STRIDE == 1) ? 1 : 0;
    constexpr int COLS = 64 * STRIDE + 2;
    constexpr int NA   = 3 * COLS * 4;       // A staging b128-groups
    constexpr int NB   = 9 * 16 * 4;         // B staging b128-groups (576)

    __shared__ __align__(16) half_t ldsA[3 * COLS * 32];
    __shared__ __align__(16) half_t ldsB[9 * 16 * 32];

    const int tid  = threadIdx.x;
    const int lane = tid & 31;
    const int wave = tid >> 5;
    const int g    = lane >> 4;
    const int ln   = lane & 15;

    const int tiles_w = Wo >> 6;
    const int ho  = blockIdx.x / tiles_w;
    const int wb  = (blockIdx.x - ho * tiles_w) * 64;   // block's first out col
    const int b   = blockIdx.y;
    const int ct  = blockIdx.z;
    const int cn  = ct * 16 + ln;

    const int yg0 = ho * STRIDE - PAD;
    const int xg0 = wb * STRIDE - PAD;

    v8f acc = {};

    for (int cc = 0; cc < Cin; cc += 32) {
        // ---- uniform per-chunk segment resolve (boundary 32-aligned) ----
        const float* sp; int segch, ci0, up;
        if (cc < c0) { sp = s0; segch = c0;       ci0 = 0;  up = u0; }
        else         { sp = s1; segch = Cin - c0; ci0 = c0; up = u1; }
        const int sH = up ? (Hi >> 1) : Hi;
        const int sW = up ? (Wi >> 1) : Wi;

        __syncthreads();   // protect previous chunk's LDS reads

        // ---- stage A: [row][col][32ch] f16, zero-padded ----
        for (int i = tid; i < NA; i += 128) {
            int pos = i >> 2;
            int cig = (i & 3) * 8;
            int r   = pos / COLS;
            int xl  = pos - r * COLS;
            int yg  = yg0 + r;
            int xg  = xg0 + xl;
            bool inb = (yg >= 0) & (yg < Hi) & (xg >= 0) & (xg < Wi);
            int ygc = min(max(yg, 0), Hi - 1);
            int xgc = min(max(xg, 0), Wi - 1);
            int ys = up ? (ygc >> 1) : ygc;
            int xs = up ? (xgc >> 1) : xgc;
            v8h pk;
            #pragma unroll
            for (int k = 0; k < 8; ++k) {
                int cloc = cc + cig + k - ci0;
                int ccl  = min(cloc, segch - 1);
                float v  = sp[((b * segch + ccl) * sH + ys) * sW + xs];
                pk[k] = (half_t)((inb && cloc < segch) ? v : 0.0f);
            }
            *(v8h*)(ldsA + pos * 32 + cig) = pk;
        }

        // ---- stage B: [tap][cout16][32ch] f16, zero-padded ----
        for (int i = tid; i < NB; i += 128) {
            int t   = i >> 6;
            int r2  = i & 63;
            int cnl = r2 >> 2;
            int kg  = (r2 & 3) * 8;
            int o   = ct * 16 + cnl;
            int ocl = min(o, Cout - 1);
            v8h pk;
            #pragma unroll
            for (int k = 0; k < 8; ++k) {
                int ci  = cc + kg + k;
                int cic = min(ci, Cin - 1);
                float w = wgt[(ocl * Cin + cic) * 9 + t];
                pk[k] = (half_t)((o < Cout && ci < Cin) ? w : 0.0f);
            }
            *(v8h*)(ldsB + (t * 16 + cnl) * 32 + kg) = pk;
        }

        __syncthreads();

        // ---- 9 taps, fully unrolled: 4 x ds_load_b128 + 1 WMMA each ----
        const int xbase = (wave * 16 + ln) * STRIDE;
        #pragma unroll
        for (int r = 0; r < 3; ++r) {
            #pragma unroll
            for (int s = 0; s < 3; ++s) {
                const int t    = r * 3 + s;
                const int aoff = (r * COLS + xbase + s) * 32 + g * 8;
                const int boff = (t * 16 + ln) * 32 + g * 8;
                v8h alo = *(const v8h*)(ldsA + aoff);
                v8h ahi = *(const v8h*)(ldsA + aoff + 16);
                v8h blo = *(const v8h*)(ldsB + boff);
                v8h bhi = *(const v8h*)(ldsB + boff + 16);
                v16h af = __builtin_shufflevector(alo, ahi,
                          0,1,2,3,4,5,6,7,8,9,10,11,12,13,14,15);
                v16h bf = __builtin_shufflevector(blo, bhi,
                          0,1,2,3,4,5,6,7,8,9,10,11,12,13,14,15);
                acc = __builtin_amdgcn_wmma_f32_16x16x32_f16(false, af, false, bf,
                                                             (short)0, acc, false, false);
            }
        }
    }

    if (cn >= Cout) return;
    const float bv = bias[cn];
    const int wo0 = wb + wave * 16 + (g ? 8 : 0);
    #pragma unroll
    for (int v = 0; v < 8; ++v) {
        float val = acc[v] + bv;
        int wo = wo0 + v;
        if (act == 1) {
            out[((b * Cout + cn) * Ho + ho) * Wo + wo] = fmaxf(val, 0.0f);
        } else if (act == 2) {
            if (cn < 2) out[((b * 2 + cn) * Ho + ho) * Wo + wo] = tanhf(val);
            else        out2[(b * Ho + ho) * Wo + wo] = 1.0f / (1.0f + __expf(-val));
        } else {
            out[((b * Cout + cn) * Ho + ho) * Wo + wo] = val;
        }
    }
}

// ---------------------------------------------------------------------------
// Host launcher
// ---------------------------------------------------------------------------
static inline void launch_conv(hipStream_t st,
                               const float* s0, int c0, int u0,
                               const float* s1, int u1,
                               const float* w, const float* bia,
                               float* out, float* out2,
                               int Cin, int Cout, int Hi, int Wi, int Ho, int Wo,
                               int stride, int act) {
    dim3 grid(Ho * (Wo >> 6), BB, (Cout + 15) / 16);
    if (stride == 1)
        conv3x3_wmma<1><<<grid, 128, 0, st>>>(s0, c0, u0, s1, u1, w, bia, out, out2,
                                              Cin, Cout, Hi, Wi, Ho, Wo, act);
    else
        conv3x3_wmma<2><<<grid, 128, 0, st>>>(s0, c0, u0, s1, u1, w, bia, out, out2,
                                              Cin, Cout, Hi, Wi, Ho, Wo, act);
}

extern "C" void kernel_launch(void* const* d_in, const int* in_sizes, int n_in,
                              void* d_out, int out_size, void* d_ws, size_t ws_size,
                              hipStream_t stream) {
    const float* Is  = (const float*)d_in[0];
    const float* Ds  = (const float*)d_in[1];
    const float* Ks  = (const float*)d_in[2];
    const float* Kt  = (const float*)d_in[3];
    const float* dT  = (const float*)d_in[4];
    const float* fw1 = (const float*)d_in[5];   const float* fb1 = (const float*)d_in[6];
    const float* fw2 = (const float*)d_in[7];   const float* fb2 = (const float*)d_in[8];
    const float* fw3 = (const float*)d_in[9];   const float* fb3 = (const float*)d_in[10];
    const float* we0 = (const float*)d_in[11];  const float* be0 = (const float*)d_in[12];
    const float* we1 = (const float*)d_in[13];  const float* be1 = (const float*)d_in[14];
    const float* we2 = (const float*)d_in[15];  const float* be2 = (const float*)d_in[16];
    const float* wd1 = (const float*)d_in[17];  const float* bd1 = (const float*)d_in[18];
    const float* wd0 = (const float*)d_in[19];  const float* bd0 = (const float*)d_in[20];
    const float* wo  = (const float*)d_in[21];  const float* bo  = (const float*)d_in[22];

    float* o = (float*)d_out;
    float* o_It = o;                       // (B,3,H,W)
    float* o_Iw = o + (size_t)3 * BHW;     // (B,3,H,W)
    float* o_V  = o + (size_t)6 * BHW;     // (B,1,H,W)
    float* o_O  = o + (size_t)7 * BHW;     // (B,1,H,W)
    float* o_gg = o + (size_t)8 * BHW;     // (B,H,W,2)
    float* o_dg = o + (size_t)10 * BHW;    // (B,2,H,W)
    float* o_zt = o + (size_t)12 * BHW;    // (B,1,H,W)

    float* ws   = (float*)d_ws;
    float* xtb  = ws;
    float* ytb  = ws + (size_t)1  * BHW;
    float* zb   = ws + (size_t)2  * BHW;
    float* zbuf = ws + (size_t)3  * BHW;   // dead after splat; reused by cin5
    float* acc  = ws + (size_t)4  * BHW;   // 4 planes; dead after normalize
    float* cin5 = ws + (size_t)3  * BHW;   // 5ch concat buffer (reuses zbuf+acc)
    float* bufA = ws + (size_t)8  * BHW;   // 32ch full-res: h1, later d0
    float* bufB = ws + (size_t)40 * BHW;   // 32ch full-res: h2, later e0
    float* e1b  = ws + (size_t)72 * BHW;   // 64ch @ H/2  = 16*BHW
    float* e2b  = ws + (size_t)88 * BHW;   // 128ch @ H/4 =  8*BHW
    float* d1b  = ws + (size_t)96 * BHW;   // 64ch @ H/2  = 16*BHW
    float* kinv = ws + (size_t)112 * BHW;  // B*9
    float* zmax = kinv + 128;              // B

    const int T = 256;
    const int GB = (BHW + T - 1) / T;

    // --- geometry + splat ---
    kinv_kernel<<<1, BB, 0, stream>>>(Ks, kinv);
    init_kernel<<<GB, T, 0, stream>>>(zbuf, acc, zmax);
    project_kernel<<<GB, T, 0, stream>>>(Ds, Kt, dT, kinv, xtb, ytb, zb, zmax, o_gg, o_zt);
    zbuf_kernel<<<GB, T, 0, stream>>>(xtb, ytb, zb, zbuf);
    splat_kernel<<<GB, T, 0, stream>>>(Is, xtb, ytb, zb, zbuf, zmax, acc);
    normalize_kernel<<<GB, T, 0, stream>>>(acc, o_Iw, o_V);

    // --- flow net: concat(Is, Ds, V) -> 32 -> 32 -> (tanh x2 | sigmoid) ---
    concat5_kernel<<<GB, T, 0, stream>>>(Is, Ds, o_V, cin5);
    launch_conv(stream, cin5, 5, 0, cin5, 0, fw1, fb1, bufA, nullptr,
                5, 32, HH, WW, HH, WW, 1, /*relu*/1);
    launch_conv(stream, bufA, 32, 0, bufA, 0, fw2, fb2, bufB, nullptr,
                32, 32, HH, WW, HH, WW, 1, 1);
    launch_conv(stream, bufB, 32, 0, bufB, 0, fw3, fb3, o_dg, o_O,
                32, 3, HH, WW, HH, WW, 1, /*flow head*/2);

    // --- U-Net ---
    concat5_kernel<<<GB, T, 0, stream>>>(o_Iw, o_V, o_O, cin5);
    launch_conv(stream, cin5, 5, 0, cin5, 0, we0, be0, bufB, nullptr,
                5, 32, HH, WW, HH, WW, 1, 1);                        // e0
    launch_conv(stream, bufB, 32, 0, bufB, 0, we1, be1, e1b, nullptr,
                32, 64, HH, WW, H2, W2, 2, 1);                       // e1
    launch_conv(stream, e1b, 64, 0, e1b, 0, we2, be2, e2b, nullptr,
                64, 128, H2, W2, H4, W4, 2, 1);                      // e2
    launch_conv(stream, e2b, 128, 1, e1b, 0, wd1, bd1, d1b, nullptr,
                192, 64, H2, W2, H2, W2, 1, 1);                      // d1 = conv(up2(e2)||e1)
    launch_conv(stream, d1b, 64, 1, bufB, 0, wd0, bd0, bufA, nullptr,
                96, 32, HH, WW, HH, WW, 1, 1);                       // d0 = conv(up2(d1)||e0)
    launch_conv(stream, bufA, 32, 0, bufA, 0, wo, bo, o_It, nullptr,
                32, 3, HH, WW, HH, WW, 1, /*none*/0);                // It
}
